// MeshRefinementModel_12945031430694
// MI455X (gfx1250) — matrix operand order repaired and checked by reading, not compiled, not run on hardware
//
#include <hip/hip_runtime.h>
#include <cmath>

// ============================================================================
// MeshRefinementModel for MI455X (gfx1250, wave32, WMMA).
//
// Activations stored node-major [N][B=8][F] (== reference's (N, B*F) view).
// All sparse ops have fixed degree with row-contiguous COO entries.
//
// d_ws layout (floats):
//   T      : 6 * S   (Chebyshev slices, S = 35709*8*32)
//   A0..A2 : 3 * S   (ping-pong activations + res-block hidden)
//   SC     : 1 * S   (shortcut)
//   CAT    : N0*8*6  (decode||refine concat)
// ============================================================================

typedef _Float16 v16h __attribute__((ext_vector_type(16)));
typedef float    v8f  __attribute__((ext_vector_type(8)));

#define BQ 8
#define KCHEB 6

static const int POOLSZ[5] = {35709, 8928, 2232, 558, 140};

struct BlkP { const float *w1, *b1, *w2, *b2, *scw, *scb; };

// ---------------------------------------------------------------------------
// dfc: out[n][b][f] = relu( emb[b,:] . dfc_w[n*32+f,:] + dfc_b[n*32+f] )
// ---------------------------------------------------------------------------
__global__ void dfc_kernel(const float* __restrict__ emb,
                           const float* __restrict__ w,
                           const float* __restrict__ bias,
                           float* __restrict__ out) {
  int t = blockIdx.x * blockDim.x + threadIdx.x;
  const int total = 140 * BQ * 32;
  if (t >= total) return;
  int f = t & 31;
  int b = (t >> 5) & 7;
  int n = t >> 8;
  int orow = n * 32 + f;
  const float* wr = w + (size_t)orow * 512;
  const float* er = emb + (size_t)b * 512;
  float acc = bias[orow];
  for (int k = 0; k < 512; ++k) acc = fmaf(er[k], wr[k], acc);
  out[t] = fmaxf(acc, 0.0f);
}

// ---------------------------------------------------------------------------
// tex [B][N][3] -> dst [N][B][3]
// ---------------------------------------------------------------------------
__global__ void tex_transpose_kernel(const float* __restrict__ tex,
                                     float* __restrict__ dst, int N) {
  unsigned t = blockIdx.x * blockDim.x + threadIdx.x;
  unsigned total = (unsigned)N * BQ * 3;
  if (t >= total) return;
  unsigned f = t % 3u;
  unsigned nb = t / 3u;
  unsigned b = nb & 7u;
  unsigned n = nb >> 3;
  dst[t] = tex[(size_t)b * N * 3 + (size_t)n * 3 + f];
}

__global__ void copy4_kernel(const float4* __restrict__ src,
                             float4* __restrict__ dst, unsigned n4) {
  unsigned t = blockIdx.x * blockDim.x + threadIdx.x;
  if (t < n4) dst[t] = src[t];
}

// ---------------------------------------------------------------------------
// Fixed-degree SpMM over [n_out, C], vectorized over float4 columns:
//   y[r,c] = scale * sum_d val[r*deg+d] * x[col[r*deg+d], c]  (- prev[r,c])
// ---------------------------------------------------------------------------
__global__ void spmm_kernel(const int* __restrict__ cols,
                            const float* __restrict__ vals, int deg,
                            const float4* __restrict__ x4,
                            float4* __restrict__ y4,
                            unsigned totalv, int Cv, int Cvshift,
                            const float4* __restrict__ prev4, float scale) {
  unsigned t = blockIdx.x * blockDim.x + threadIdx.x;
  if (t >= totalv) return;
  unsigned r, c;
  if (Cvshift >= 0) { r = t >> Cvshift; c = t & (unsigned)(Cv - 1); }
  else             { r = t / (unsigned)Cv; c = t - r * (unsigned)Cv; }
  const int*   cp = cols + (size_t)r * deg;
  const float* vp = vals + (size_t)r * deg;
  float4 acc = make_float4(0.f, 0.f, 0.f, 0.f);
  for (int d = 0; d < deg; ++d) {
    float  v  = vp[d];
    float4 xv = x4[(size_t)cp[d] * Cv + c];
    acc.x = fmaf(v, xv.x, acc.x);
    acc.y = fmaf(v, xv.y, acc.y);
    acc.z = fmaf(v, xv.z, acc.z);
    acc.w = fmaf(v, xv.w, acc.w);
  }
  acc.x *= scale; acc.y *= scale; acc.z *= scale; acc.w *= scale;
  if (prev4) {
    float4 pv = prev4[t];
    acc.x -= pv.x; acc.y -= pv.y; acc.z -= pv.z; acc.w -= pv.w;
  }
  y4[t] = acc;
}

// ---------------------------------------------------------------------------
// Chebyshev combine GEMM via v_wmma_f32_16x16x32_f16, templated on FIN so
// Kd/KSTEPS are compile-time; FIN=16/32 use b128 fragment loads with zero
// per-element address math; FIN=3/6 use branchless clamp+select gathers
// (constant-divisor division).  One wave -> 16x16 C tile, 4 waves/block.
// Fused bias / residual / activation / strided-store epilogue.
// act: 0=none, 1=relu, 2=tanh.
// ---------------------------------------------------------------------------
__device__ __forceinline__ void put4(v16h& v, int i, float4 f) {
  v[i]     = (_Float16)f.x;
  v[i + 1] = (_Float16)f.y;
  v[i + 2] = (_Float16)f.z;
  v[i + 3] = (_Float16)f.w;
}

template <int FIN>
__global__ void cheb_gemm_kernel(const float* __restrict__ T, size_t S,
                                 int M, int Fout,
                                 const float* __restrict__ W,
                                 const float* __restrict__ bias,
                                 const float* __restrict__ addsrc, int act,
                                 float* __restrict__ out,
                                 int onstr, int obstr, int ofoff) {
  constexpr int  Kd      = KCHEB * FIN;
  constexpr int  KSTEPS  = (Kd + 31) / 32;
  constexpr bool ALIGNED = (FIN % 8 == 0);  // FIN = 16, 32

  int lane = threadIdx.x & 31;
  int wave = threadIdx.x >> 5;
  int row0 = (blockIdx.x * 4 + wave) * 16;
  int lr   = lane & 15;
  int half = lane >> 4;

  // A addressing (clamped; padded rows never stored)
  int row  = row0 + lr;
  int rowc = row < M ? row : M - 1;
  const float* Abase =
      T + (size_t)(rowc >> 3) * (BQ * FIN) + (size_t)(rowc & 7) * FIN;
  int khiA = half << 3;   // A frag: +8  K-offset for upper half-wave
  int khiB = half << 3;   // B frag: +8  K-offset for upper half-wave

  // B addressing (W is [Fout, Kd] row-major; tile column = lr)
  int fo_g = blockIdx.y * 16 + lr;
  int foc  = fo_g < Fout ? fo_g : Fout - 1;
  const float* Wrow = W + (size_t)foc * Kd;

  v8f acc = {};
#pragma unroll
  for (int ks = 0; ks < KSTEPS; ++ks) {
    int k0 = ks << 5;
    v16h av, bv;
    if (ALIGNED) {
      // Each 8-element run is contiguous and within one Chebyshev slice.
      int base0 = k0 + khiA;        // multiple of 8
      int base1 = base0 + 16;
      const float4* pa0 = (const float4*)(Abase +
          (size_t)(base0 / FIN) * S + (base0 % FIN));
      const float4* pa1 = (const float4*)(Abase +
          (size_t)(base1 / FIN) * S + (base1 % FIN));
      put4(av, 0,  pa0[0]); put4(av, 4,  pa0[1]);
      put4(av, 8,  pa1[0]); put4(av, 12, pa1[1]);
      const float4* pb = (const float4*)(Wrow + k0 + khiB);  // two 8-runs, +16 apart
      put4(bv, 0,  pb[0]);  put4(bv, 4,  pb[1]);
      put4(bv, 8,  pb[4]);  put4(bv, 12, pb[5]);
    } else {
#pragma unroll
      for (int t = 0; t < 16; ++t) {
        int ja  = k0 + ((t & 8) << 1) + khiA + (t & 7);
        int jac = ja < Kd ? ja : Kd - 1;     // clamp (keep loads in-bounds)
        int kc  = jac / FIN;                 // constant divisor -> mul/shift
        int fi  = jac - kc * FIN;
        float a = Abase[(size_t)kc * S + fi];
        a = (ja < Kd) ? a : 0.f;             // branchless zero-pad
        int jb  = k0 + ((t & 8) << 1) + khiB + (t & 7);
        int jbc = jb < Kd ? jb : Kd - 1;
        float w = Wrow[jbc];
        w = (jb < Kd) ? w : 0.f;
        av[t] = (_Float16)a;
        bv[t] = (_Float16)w;
      }
    }
    acc = __builtin_amdgcn_wmma_f32_16x16x32_f16(
        /*neg_a=*/false, av, /*neg_b=*/false, bv,
        /*c_mod=*/(short)0, acc, /*reuse_a=*/false, /*reuse_b=*/false);
  }

  // C/D layout: lanes 0-15 -> N=lane, M=i ; lanes 16-31 -> N=lane-16, M=8+i
#pragma unroll
  for (int i = 0; i < 8; ++i) {
    int m    = i + (half << 3);
    int grow = row0 + m;
    if (grow < M && fo_g < Fout) {
      float v = acc[i] + bias[fo_g];
      if (addsrc) v += addsrc[(size_t)grow * Fout + fo_g];
      if (act == 1)      v = fmaxf(v, 0.f);
      else if (act == 2) v = tanhf(v);
      int gn = grow >> 3;
      int gb = grow & 7;
      out[(size_t)gn * onstr + (size_t)gb * obstr + ofoff + fo_g] = v;
    }
  }
}

// ===================== host-side orchestration helpers ======================

static inline int ilog2_if_pow2(int x) {
  return (x > 0 && (x & (x - 1)) == 0) ? __builtin_ctz(x) : -1;
}

static void launch_spmm(hipStream_t s, const int* cols, const float* vals,
                        int deg, const float* x, float* y, int n_out, int C,
                        const float* prev, float scale) {
  int Cv = C >> 2;  // C is always a multiple of 4 (C = 8*F, F in {3,6,16,32})
  unsigned totalv = (unsigned)n_out * (unsigned)Cv;
  int cs = ilog2_if_pow2(Cv);
  spmm_kernel<<<(totalv + 255u) / 256u, 256, 0, s>>>(
      cols, vals, deg, (const float4*)x, (float4*)y, totalv, Cv, cs,
      (const float4*)prev, scale);
}

static void launch_gemm(hipStream_t s, const float* T, size_t S, int M,
                        int Fin, int Fout, const float* W, const float* bias,
                        const float* addsrc, int act, float* out,
                        int onstr, int obstr, int ofoff) {
  int mtiles = (M + 15) / 16;
  dim3 grid((mtiles + 3) / 4, (Fout + 15) / 16);
  switch (Fin) {
    case 3:
      cheb_gemm_kernel<3><<<grid, 128, 0, s>>>(T, S, M, Fout, W, bias, addsrc,
                                               act, out, onstr, obstr, ofoff);
      break;
    case 6:
      cheb_gemm_kernel<6><<<grid, 128, 0, s>>>(T, S, M, Fout, W, bias, addsrc,
                                               act, out, onstr, obstr, ofoff);
      break;
    case 16:
      cheb_gemm_kernel<16><<<grid, 128, 0, s>>>(T, S, M, Fout, W, bias, addsrc,
                                                act, out, onstr, obstr, ofoff);
      break;
    default:
      cheb_gemm_kernel<32><<<grid, 128, 0, s>>>(T, S, M, Fout, W, bias, addsrc,
                                                act, out, onstr, obstr, ofoff);
      break;
  }
}

static void cheb_conv(hipStream_t s, const float* X, int N, int Fin, int Fout,
                      const int* lcols, const float* lvals, int deg,
                      const float* W, const float* bias,
                      const float* addsrc, int act,
                      float* T, size_t S,
                      float* out, int onstr, int obstr, int ofoff) {
  int C = BQ * Fin;
  unsigned e4 = (unsigned)N * (unsigned)C / 4u;
  copy4_kernel<<<(e4 + 255u) / 256u, 256, 0, s>>>((const float4*)X,
                                                  (float4*)T, e4);      // T0
  launch_spmm(s, lcols, lvals, deg, T, T + S, N, C, nullptr, 1.f);      // T1
  for (int k = 2; k < KCHEB; ++k)                                       // Tk
    launch_spmm(s, lcols, lvals, deg, T + (size_t)(k - 1) * S,
                T + (size_t)k * S, N, C, T + (size_t)(k - 2) * S, 2.f);
  launch_gemm(s, T, S, N * BQ, Fin, Fout, W, bias, addsrc, act,
              out, onstr, obstr, ofoff);
}

static void res_block(hipStream_t s, const float* X, int N, int Fin, int Fout,
                      const int* lcols, const float* lvals,
                      const BlkP& bp, bool has_sc,
                      float* T, size_t S, float* H, float* SCb,
                      float* out, int onstr, int obstr, int ofoff) {
  const float* scsrc = X;  // identity shortcut when Fin == Fout
  if (has_sc) {
    cheb_conv(s, X, N, Fin, Fout, lcols, lvals, 7, bp.scw, bp.scb,
              nullptr, /*act*/0, T, S, SCb, BQ * Fout, Fout, 0);
    scsrc = SCb;
  }
  cheb_conv(s, X, N, Fin, Fout, lcols, lvals, 7, bp.w1, bp.b1,
            nullptr, /*relu*/1, T, S, H, BQ * Fout, Fout, 0);
  cheb_conv(s, H, N, Fout, Fout, lcols, lvals, 7, bp.w2, bp.b2,
            scsrc, /*relu*/1, T, S, out, onstr, obstr, ofoff);
}

// ============================== entry point =================================

extern "C" void kernel_launch(void* const* d_in, const int* in_sizes, int n_in,
                              void* d_out, int out_size, void* d_ws,
                              size_t ws_size, hipStream_t stream) {
  (void)in_sizes; (void)n_in; (void)out_size; (void)ws_size;

  // ---- unpack inputs (setup_inputs insertion order, recursive flatten) ----
  int p = 0;
  const float* image_emb = (const float*)d_in[p++];
  const float* tex       = (const float*)d_in[p++];
  const float* dfc_w     = (const float*)d_in[p++];
  const float* dfc_b     = (const float*)d_in[p++];

  const bool dec_sc[4] = {true, false, false, true};   // DEC_F fin!=fout
  const bool ref_sc[5] = {true, true, false, true, true};
  BlkP dec[4], ref[5];
  for (int i = 0; i < 4; ++i) {
    dec[i].w1 = (const float*)d_in[p++]; dec[i].b1 = (const float*)d_in[p++];
    dec[i].w2 = (const float*)d_in[p++]; dec[i].b2 = (const float*)d_in[p++];
    if (dec_sc[i]) { dec[i].scw = (const float*)d_in[p++];
                     dec[i].scb = (const float*)d_in[p++]; }
    else           { dec[i].scw = nullptr; dec[i].scb = nullptr; }
  }
  for (int i = 0; i < 5; ++i) {
    ref[i].w1 = (const float*)d_in[p++]; ref[i].b1 = (const float*)d_in[p++];
    ref[i].w2 = (const float*)d_in[p++]; ref[i].b2 = (const float*)d_in[p++];
    if (ref_sc[i]) { ref[i].scw = (const float*)d_in[p++];
                     ref[i].scb = (const float*)d_in[p++]; }
    else           { ref[i].scw = nullptr; ref[i].scb = nullptr; }
  }
  const float* comb_w = (const float*)d_in[p++];
  const float* comb_b = (const float*)d_in[p++];

  const int* lap_cols[5]; const float* lap_val[5];
  const int* up_cols[4];  const float* up_val[4];
  for (int i = 0; i < 5; ++i) {  // idx arrays are [2, nnz]; cols = row 1
    const int* idx = (const int*)d_in[p++];
    lap_cols[i] = idx + (size_t)POOLSZ[i] * 7;
  }
  for (int i = 0; i < 5; ++i) lap_val[i] = (const float*)d_in[p++];
  for (int i = 0; i < 4; ++i) {
    const int* idx = (const int*)d_in[p++];
    up_cols[i] = idx + (size_t)POOLSZ[i] * 3;
  }
  for (int i = 0; i < 4; ++i) up_val[i] = (const float*)d_in[p++];
  const int* down_idx = (const int*)d_in[p++];
  const int* down_cols = down_idx + (size_t)POOLSZ[1] * 3;
  const float* down_val = (const float*)d_in[p++];

  // ---- workspace carve-up ----
  float* ws = (float*)d_ws;
  const size_t S = (size_t)35709 * BQ * 32;  // max per-slice elements
  float* T   = ws;
  float* A0  = ws + 6 * S;
  float* A1  = A0 + S;
  float* A2  = A1 + S;
  float* SCb = A2 + S;
  float* CAT = SCb + S;  // [N0][B][6]
  const int N0 = POOLSZ[0], N1 = POOLSZ[1];

  // ======================= mesh decoder =======================
  dfc_kernel<<<(140 * BQ * 32 + 255) / 256, 256, 0, stream>>>(image_emb, dfc_w,
                                                              dfc_b, A0);
  const int DEC_F[5] = {32, 16, 16, 16, 3};
  const int lap_order[4] = {3, 2, 1, 0};
  for (int i = 0; i < 4; ++i) {
    int li = lap_order[i];
    int N = POOLSZ[li];
    int Fin = DEC_F[i], Fout = DEC_F[i + 1];
    // upsample POOL[li+1] -> POOL[li]
    launch_spmm(stream, up_cols[li], up_val[li], 3, A0, A1, N, BQ * Fin,
                nullptr, 1.f);
    bool last = (i == 3);  // write decode_color straight into CAT[..,0:3]
    res_block(stream, A1, N, Fin, Fout, lap_cols[li], lap_val[li], dec[i],
              dec_sc[i], T, S, A2, SCb,
              last ? CAT : A0, last ? BQ * 6 : BQ * Fout, last ? 6 : Fout, 0);
  }

  // ======================= mesh refiner =======================
  {
    unsigned tt = (unsigned)N0 * BQ * 3;
    tex_transpose_kernel<<<(tt + 255u) / 256u, 256, 0, stream>>>(tex, A1, N0);
  }
  res_block(stream, A1, N0, 3, 16, lap_cols[0], lap_val[0], ref[0], true,
            T, S, A2, SCb, A0, BQ * 16, 16, 0);
  launch_spmm(stream, down_cols, down_val, 3, A0, A1, N1, BQ * 16, nullptr, 1.f);
  res_block(stream, A1, N1, 16, 32, lap_cols[1], lap_val[1], ref[1], true,
            T, S, A2, SCb, A0, BQ * 32, 32, 0);
  res_block(stream, A0, N1, 32, 32, lap_cols[1], lap_val[1], ref[2], false,
            T, S, A2, SCb, A1, BQ * 32, 32, 0);
  launch_spmm(stream, up_cols[0], up_val[0], 3, A1, A0, N0, BQ * 32, nullptr, 1.f);
  res_block(stream, A0, N0, 32, 16, lap_cols[0], lap_val[0], ref[3], true,
            T, S, A2, SCb, A1, BQ * 16, 16, 0);
  // refine_color straight into CAT[..,3:6]
  res_block(stream, A1, N0, 16, 3, lap_cols[0], lap_val[0], ref[4], true,
            T, S, A2, SCb, CAT, BQ * 6, 6, 3);

  // ======================= combine + tanh =======================
  // d_out is [B][N0][3]: node stride 3, batch stride N0*3.
  cheb_conv(stream, CAT, N0, 6, 3, lap_cols[0], lap_val[0], 7, comb_w, comb_b,
            nullptr, /*tanh*/2, T, S, (float*)d_out, 3, N0 * 3, 0);
}
